// SpatialSelfAttention_28278064677103
// MI455X (gfx1250) — matrix operand order repaired and checked by reading, not compiled, not run on hardware
//
#include <hip/hip_runtime.h>
#include <hip/hip_bf16.h>

typedef __attribute__((ext_vector_type(16))) _Float16 v16h;
typedef __attribute__((ext_vector_type(8)))  float    v8f;
typedef __attribute__((ext_vector_type(4)))  unsigned int v4u;
typedef __attribute__((ext_vector_type(8)))  int      v8i;
typedef __attribute__((ext_vector_type(4)))  int      v4i;

#define HW    16384   // 128*128
#define WIDTH 128

// A-operand k order used here (16x32 MxK, contiguous, B matches): lane m (0-15) holds K={0..15},
// lane m+16 holds K={16..31}; half j of the v16h maps to this channel:
__device__ __forceinline__ int chA(int j, int hi) {
    return hi * 16 + j;
}

// ---------------------------------------------------------------------------
// Kernel 1: windowed cosine attention. grid=(1024 windows, 4 head-pairs),
// block=64 threads (2 waves, wave w -> head blockIdx.y*2+w).
// ---------------------------------------------------------------------------
__global__ __launch_bounds__(64) void attn_kernel(
    const float* __restrict__ qkv, const float* __restrict__ ch,
    const float* __restrict__ logit_scale,
    const float* __restrict__ rel_bias_table,
    const int*   __restrict__ rel_index,
    _Float16*    __restrict__ xws)
{
    __shared__ unsigned short s_idx[64 * 64];      // 8192 B (shared across heads)
    __shared__ float          s_tab[225 * 8];      // 7200 B
    __shared__ float          s_chm[64 * 32];      // 8192 B (head-independent gate)
    __shared__ _Float16       s_q[2][64 * 32];     // 8192 B
    __shared__ _Float16       s_k[2][64 * 32];     // 8192 B
    __shared__ _Float16       s_v[2][64 * 32];     // 8192 B
    __shared__ _Float16       s_p[2][64 * 64];     // 16384 B   total = 64544 B

    const int tid  = threadIdx.x;
    const int wave = tid >> 5;
    const int lane = tid & 31;
    const int lo = lane & 15, hi = lane >> 4;

    const int wid = blockIdx.x;                    // 0..1023
    const int b   = wid >> 8;
    const int wy  = (wid >> 4) & 15;
    const int wx  = wid & 15;
    const int l   = blockIdx.y * 2 + wave;         // head 0..7

    // ---- phase 0: block-wide shared tables + channel-helper gate ----
    for (int i = tid; i < 225 * 8; i += 64) s_tab[i] = rel_bias_table[i];
    for (int i = tid; i < 4096;    i += 64) s_idx[i] = (unsigned short)rel_index[i];
    for (int e = tid; e < 2048; e += 64) {         // chm = mean over 8 groups
        int c = e >> 6, t = e & 63;
        int y = wy * 8 + (t >> 3), x = wx * 8 + (t & 7);
        const float* cp = ch + ((size_t)b * 256 + c) * HW + y * WIDTH + x;
        float s = 0.f;
        #pragma unroll
        for (int g = 0; g < 8; ++g) s += cp[(size_t)g * 32 * HW];
        s_chm[t * 32 + c] = s * 0.125f;
    }
    __syncthreads();

    _Float16* q = s_q[wave];
    _Float16* k = s_k[wave];
    _Float16* v = s_v[wave];
    _Float16* p = s_p[wave];

    // ---- phase 1: load q,k,v (v gated by chm) for this head ----
    const size_t qb = (((size_t)b * 8 + l) * 96) * HW;
    for (int e = lane; e < 2048; e += 32) {
        int c = e >> 6, t = e & 63;
        int y = wy * 8 + (t >> 3), x = wx * 8 + (t & 7);
        size_t off = (size_t)c * HW + (size_t)y * WIDTH + x;
        q[t * 32 + c] = (_Float16)qkv[qb + off];
        k[t * 32 + c] = (_Float16)qkv[qb + (size_t)32 * HW + off];
        v[t * 32 + c] = (_Float16)(qkv[qb + (size_t)64 * HW + off] * s_chm[t * 32 + c]);
    }
    // l2-normalize q,k per token
    for (int t = lane; t < 64; t += 32) {
        float sq = 0.f, sk = 0.f;
        #pragma unroll
        for (int c = 0; c < 32; ++c) {
            float a_ = (float)q[t * 32 + c]; sq += a_ * a_;
            float b_ = (float)k[t * 32 + c]; sk += b_ * b_;
        }
        float iq = 1.f / fmaxf(sqrtf(sq), 1e-12f);
        float ik = 1.f / fmaxf(sqrtf(sk), 1e-12f);
        #pragma unroll
        for (int c = 0; c < 32; ++c) {
            q[t * 32 + c] = (_Float16)((float)q[t * 32 + c] * iq);
            k[t * 32 + c] = (_Float16)((float)k[t * 32 + c] * ik);
        }
    }

    const float scale = __expf(fminf(logit_scale[l], 4.605170185988092f)); // log(100)

    // ---- phase 2: S = Qn . Kn^T  (64x64 via 4x4 WMMA tiles, K=32) ----
    v16h bk[4];                                    // B frags: column tile = key tokens
    #pragma unroll
    for (int tc = 0; tc < 4; ++tc)
        #pragma unroll
        for (int j = 0; j < 16; ++j)
            bk[tc][j] = k[(tc * 16 + lo) * 32 + 16 * hi + j];

    #pragma unroll
    for (int tm = 0; tm < 4; ++tm) {
        v16h a;
        #pragma unroll
        for (int j = 0; j < 16; ++j)
            a[j] = q[(tm * 16 + lo) * 32 + chA(j, hi)];
        v8f s[4];
        #pragma unroll
        for (int tc = 0; tc < 4; ++tc) {
            v8f z = {};
            s[tc] = __builtin_amdgcn_wmma_f32_16x16x32_f16(
                        false, a, false, bk[tc], (short)0, z, false, false);
        }
        // ---- fused scale + rel-pos bias + row softmax (rows span 16 lanes) ----
        #pragma unroll
        for (int r = 0; r < 8; ++r) {
            int m = tm * 16 + r + 8 * hi;
            float val[4];
            #pragma unroll
            for (int tc = 0; tc < 4; ++tc) {
                int n = tc * 16 + lo;
                float bias = s_tab[(int)s_idx[m * 64 + n] * 8 + l];
                val[tc] = s[tc][r] * scale + bias;
            }
            float mx = fmaxf(fmaxf(val[0], val[1]), fmaxf(val[2], val[3]));
            #pragma unroll
            for (int off = 1; off < 16; off <<= 1)
                mx = fmaxf(mx, __shfl_xor(mx, off, 32));
            float sum = 0.f;
            #pragma unroll
            for (int tc = 0; tc < 4; ++tc) { val[tc] = __expf(val[tc] - mx); sum += val[tc]; }
            #pragma unroll
            for (int off = 1; off < 16; off <<= 1)
                sum += __shfl_xor(sum, off, 32);
            float inv = 1.f / sum;
            #pragma unroll
            for (int tc = 0; tc < 4; ++tc)
                p[m * 64 + tc * 16 + lo] = (_Float16)(val[tc] * inv);
        }
    }

    // ---- phase 3: out = P @ Vg  (64x32, K=64 -> two WMMA chains) ----
    v16h bv[2][2];
    #pragma unroll
    for (int kc = 0; kc < 2; ++kc)
        #pragma unroll
        for (int nc = 0; nc < 2; ++nc)
            #pragma unroll
            for (int j = 0; j < 16; ++j)
                bv[kc][nc][j] = v[(kc * 32 + 16 * hi + j) * 32 + nc * 16 + lo];

    #pragma unroll
    for (int tm = 0; tm < 4; ++tm) {
        v16h a0, a1;
        #pragma unroll
        for (int j = 0; j < 16; ++j) {
            a0[j] = p[(tm * 16 + lo) * 64 + chA(j, hi)];
            a1[j] = p[(tm * 16 + lo) * 64 + 32 + chA(j, hi)];
        }
        #pragma unroll
        for (int nc = 0; nc < 2; ++nc) {
            v8f d = {};
            d = __builtin_amdgcn_wmma_f32_16x16x32_f16(false, a0, false, bv[0][nc], (short)0, d, false, false);
            d = __builtin_amdgcn_wmma_f32_16x16x32_f16(false, a1, false, bv[1][nc], (short)0, d, false, false);
            #pragma unroll
            for (int r = 0; r < 8; ++r) {
                int m = tm * 16 + r + 8 * hi;                 // token in window
                int y = wy * 8 + (m >> 3), x = wx * 8 + (m & 7);
                size_t pos = ((size_t)b * 128 + y) * 128 + x; // position-major scratch
                xws[pos * 256 + l * 32 + nc * 16 + lo] = (_Float16)d[r];
            }
        }
    }
}

// ---------------------------------------------------------------------------
// Kernel 2: 1x1 conv = GEMM (65536 x 256) @ W^T (256 x 256) + bias.
// grid=(1024 pos-tiles, 4 out-channel tiles), block=128 (4 waves).
// X tile (contiguous 32KB) is pulled global->LDS by the Tensor Data Mover.
// ---------------------------------------------------------------------------
__global__ __launch_bounds__(128) void proj_kernel(
    const _Float16* __restrict__ xws,
    const float*    __restrict__ proj_w,
    const float*    __restrict__ proj_b,
    float*          __restrict__ out)
{
    __shared__ __align__(16) _Float16 s_x[64 * 256];   // 32 KB (aliased as f32 out-stage)
    __shared__ __align__(16) _Float16 s_w[64 * 256];   // 32 KB

    const int tid  = threadIdx.x;
    const int wv   = tid >> 5;
    const int lane = tid & 31;
    const int lo = lane & 15, hi = lane >> 4;
    const size_t p0 = (size_t)blockIdx.x * 64;         // 64 consecutive positions (same row)
    const int o0 = blockIdx.y * 64;

#if defined(__has_builtin) && __has_builtin(__builtin_amdgcn_tensor_load_to_lds) && \
    __has_builtin(__builtin_amdgcn_s_wait_tensorcnt)
    // ---- TDM async copy: 2-D tile, 256 f16 per row x 64 rows, row stride 256 ----
    if (wv == 0) {
        unsigned long long ga = (unsigned long long)(const void*)(xws + p0 * 256);
        unsigned int lds = (unsigned int)(unsigned long long)(const void*)s_x; // low 32b = LDS offset
        v4u g0; // D# group0: count=1, lds_addr, global_addr[56:0], type=2
        g0[0] = 1u;
        g0[1] = lds;
        g0[2] = (unsigned int)ga;
        g0[3] = (unsigned int)((ga >> 32) & 0x01FFFFFFu) | (2u << 30);
        v8i g1; // D# group1: data_size=2B, tensor 256 x 65536, tile 256 x 64, stride 256
        g1[0] = 0x00010000;              // data_size=1 (2 bytes), no mask/pad/iterate
        g1[1] = (int)(256u << 16);       // tensor_dim0 = 256 (bits 79:48)
        g1[2] = 0;                       // tensor_dim0 hi, tensor_dim1 lo
        g1[3] = (int)((256u << 16) | 1); // tensor_dim1 = 65536, tile_dim0 = 256
        g1[4] = 64;                      // tile_dim1 = 64
        g1[5] = 256;                     // tensor_dim0_stride = 256 (bits 207:160)
        g1[6] = 0;
        g1[7] = 0;
        v4i z4 = {};
        v8i z8 = {};
        // 6-arg form (clang-23 / therock-10.0 headers): groups 0..3 + extra group + cpol
        __builtin_amdgcn_tensor_load_to_lds(g0, g1, z4, z4, z8, 0);
        __builtin_amdgcn_s_wait_tensorcnt(0);
    }
#else
    {   // fallback: vectorized cooperative copy
        const uint4* src = (const uint4*)(xws + p0 * 256);
        uint4* dst = (uint4*)s_x;
        for (int i = tid; i < 2048; i += 128) dst[i] = src[i];
    }
#endif
    for (int i = tid; i < 16384; i += 128) {           // W tile f32 -> f16 (overlaps DMA)
        int o = i >> 8, c = i & 255;
        s_w[i] = (_Float16)proj_w[(size_t)(o0 + o) * 256 + c];
    }
    __syncthreads();                                   // publishes TDM-filled s_x to all waves

    v8f d[4] = {};                                     // wave wv owns N cols [wv*16, wv*16+16)
    #pragma unroll
    for (int kc = 0; kc < 8; ++kc) {
        v16h bf;
        #pragma unroll
        for (int j = 0; j < 16; ++j)
            bf[j] = s_w[(wv * 16 + lo) * 256 + kc * 32 + 16 * hi + j];
        #pragma unroll
        for (int tm = 0; tm < 4; ++tm) {
            v16h a;
            #pragma unroll
            for (int j = 0; j < 16; ++j)
                a[j] = s_x[(tm * 16 + lo) * 256 + kc * 32 + chA(j, hi)];
            d[tm] = __builtin_amdgcn_wmma_f32_16x16x32_f16(
                        false, a, false, bf, (short)0, d[tm], false, false);
        }
    }
    __syncthreads();                                   // all waves done reading s_x

    float* s_o = (float*)s_x;                          // reuse as 64x64 f32 out-stage
    #pragma unroll
    for (int tm = 0; tm < 4; ++tm)
        #pragma unroll
        for (int r = 0; r < 8; ++r) {
            int m = tm * 16 + r + 8 * hi;
            s_o[m * 64 + wv * 16 + lo] = d[tm][r];
        }
    __syncthreads();

    // coalesced channel-major store: 64 consecutive x per output channel row
    for (int i = tid; i < 4096; i += 128) {
        int o = i >> 6, m = i & 63;
        size_t pos = p0 + m;
        int bb  = (int)(pos >> 14);
        int rem = (int)(pos & 16383);
        out[((size_t)bb * 256 + o0 + o) * HW + rem] = s_o[i] + proj_b[o0 + o];
    }
}

extern "C" void kernel_launch(void* const* d_in, const int* in_sizes, int n_in,
                              void* d_out, int out_size, void* d_ws, size_t ws_size,
                              hipStream_t stream) {
    const float* qkv = (const float*)d_in[0];
    const float* ch  = (const float*)d_in[1];
    const float* ls  = (const float*)d_in[2];
    const float* tab = (const float*)d_in[3];
    const float* pw  = (const float*)d_in[4];
    const float* pb  = (const float*)d_in[5];
    const int*   ri  = (const int*)d_in[6];
    float* out = (float*)d_out;
    _Float16* xws = (_Float16*)d_ws;   // 4*128*128*256 f16 = 32 MB position-major scratch

    attn_kernel<<<dim3(1024, 4), dim3(64), 0, stream>>>(qkv, ch, ls, tab, ri, xws);
    proj_kernel<<<dim3(1024, 4), dim3(128), 0, stream>>>(xws, pw, pb, out);
}
